// TemporalMemoryModule_47665547051320
// MI455X (gfx1250) — matrix operand order repaired and checked by reading, not compiled, not run on hardware
//
#include <hip/hip_runtime.h>
#include <hip/hip_bf16.h>
#include <math.h>

// ---------------- problem constants (fixed by the reference) ----------------
#define F_DIM   256
#define M_SLOTS 10
#define B_ROWS  32768
#define DECAYF  0.9f

typedef __attribute__((ext_vector_type(16))) __bf16 v16bf;
typedef __attribute__((ext_vector_type(8)))  float  v8f;

// fp32 -> bf16 round-to-nearest-even (prep kernels and bf16 kernel outputs)
__device__ __forceinline__ __bf16 f2bf(float x) {
  unsigned u = __builtin_bit_cast(unsigned, x);
  u = u + 0x7FFFu + ((u >> 16) & 1u);
  unsigned short h = (unsigned short)(u >> 16);
  return __builtin_bit_cast(__bf16, h);
}

// ---------------- bf16 fragment loaders (pure b128 loads, no conversion) ----------------
// A fragment (16x32): lane half hl owns K = {hl*8..hl*8+7} U {16+hl*8..16+hl*8+7}
__device__ __forceinline__ v16bf load_a_frag(const __bf16* __restrict__ rowp, int k0, int hl) {
  union { v16bf v; uint4 q[2]; } u;
  u.q[0] = *(const uint4*)(rowp + k0 + hl * 8);
  u.q[1] = *(const uint4*)(rowp + k0 + 16 + hl * 8);
  return u.v;
}
// B fragment (32x16): lane holds column n = lane%16; half hl owns K = {hl*8..hl*8+7} U {16+hl*8..16+hl*8+7}
// W is bf16 row-major (N x K): element (n,k) at W[n*F + k]
__device__ __forceinline__ v16bf load_b_frag(const __bf16* __restrict__ W, int n, int k0, int hl) {
  const __bf16* p = W + (size_t)n * F_DIM + k0 + hl * 8;
  union { v16bf v; uint4 q[2]; } u;
  u.q[0] = *(const uint4*)p;
  u.q[1] = *(const uint4*)(p + 16);
  return u.v;
}

__device__ __forceinline__ v8f wmma_bf16(v16bf a, v16bf b, v8f c) {
  return __builtin_amdgcn_wmma_f32_16x16x32_bf16(false, a, false, b, (short)0, c, false, false);
}

__device__ __forceinline__ float sigmoidf_(float x) { return 1.0f / (1.0f + __expf(-x)); }

// ---------------- prep: fp32 -> bf16 bulk convert (grid-stride, float4 in / 4xbf16 out) ----
__global__ __launch_bounds__(256) void cvt_kernel(const float* __restrict__ src,
                                                  __bf16* __restrict__ dst, int n4) {
  for (int i = blockIdx.x * blockDim.x + threadIdx.x; i < n4; i += gridDim.x * blockDim.x) {
    float4 f = ((const float4*)src)[i];
    union { __bf16 h[4]; uint2 u; } o;
    o.h[0] = f2bf(f.x); o.h[1] = f2bf(f.y); o.h[2] = f2bf(f.z); o.h[3] = f2bf(f.w);
    ((uint2*)dst)[i] = o.u;
  }
}

// prep: gather hidden_states[idx] rows and convert -> dense (B,F) bf16
__global__ __launch_bounds__(256) void gather_cvt_kernel(const float* __restrict__ hs,
                                                         const int* __restrict__ idx,
                                                         __bf16* __restrict__ dst) {
  int t = blockIdx.x * 256 + threadIdx.x;          // over B*F/4
  int b = t >> 6;                                  // F/4 = 64 quads per row
  int j4 = (t & 63) << 2;
  float4 f = *(const float4*)(hs + (size_t)idx[b] * F_DIM + j4);
  union { __bf16 h[4]; uint2 u; } o;
  o.h[0] = f2bf(f.x); o.h[1] = f2bf(f.y); o.h[2] = f2bf(f.z); o.h[3] = f2bf(f.w);
  ((uint2*)dst)[t] = o.u;
}

// prep: WkT[n,k] = bf16(Wk[k,n])  (so the qk GEMM's B operand is contiguous too)
__global__ __launch_bounds__(256) void tpose_cvt_kernel(const float* __restrict__ Wk,
                                                        __bf16* __restrict__ dst) {
  int t = blockIdx.x * 256 + threadIdx.x;          // F*F threads
  int n = t >> 8, k = t & 255;
  dst[t] = f2bf(Wk[(size_t)k * F_DIM + n]);
}

// ---------------- K1: fused LSTM gates GEMM + nonlinearity -> h_new (bf16) ----------------
// One wave per 16x16 h_new patch; wave owns the same 16-col slice of all four gates.
__global__ __launch_bounds__(256) void lstm_kernel(
    const int* __restrict__ idx, const __bf16* __restrict__ xb,
    const __bf16* __restrict__ hgb, const float* __restrict__ cs,
    const __bf16* __restrict__ Wih, const __bf16* __restrict__ Whh,
    const float* __restrict__ b_ih, const float* __restrict__ b_hh,
    __bf16* __restrict__ h_new) {
  const int lane  = threadIdx.x & 31;
  const int wave  = threadIdx.x >> 5;
  const int patch = blockIdx.x * 8 + wave;
  const int NT = F_DIM / 16;
  const int m0 = (patch / NT) * 16;
  const int n0 = (patch % NT) * 16;
  const int r  = lane & 15, hl = lane >> 4;

  const __bf16* xrow = xb  + (size_t)(m0 + r) * F_DIM;
  const __bf16* hrow = hgb + (size_t)(m0 + r) * F_DIM;

  v8f acc[4];
#pragma unroll
  for (int g = 0; g < 4; ++g) acc[g] = v8f{};

  for (int k0 = 0; k0 < F_DIM; k0 += 32) {
    v16bf ax = load_a_frag(xrow, k0, hl);
    v16bf ah = load_a_frag(hrow, k0, hl);
#pragma unroll
    for (int g = 0; g < 4; ++g) {
      v16bf bi = load_b_frag(Wih, g * F_DIM + n0 + r, k0, hl);
      acc[g] = wmma_bf16(ax, bi, acc[g]);
      v16bf bh = load_b_frag(Whh, g * F_DIM + n0 + r, k0, hl);
      acc[g] = wmma_bf16(ah, bh, acc[g]);
    }
  }

  const int col = n0 + r;                    // C layout: N = lane%16
  float bias[4];
#pragma unroll
  for (int g = 0; g < 4; ++g) bias[g] = b_ih[g * F_DIM + col] + b_hh[g * F_DIM + col];
#pragma unroll
  for (int v = 0; v < 8; ++v) {              // C layout: M = v + 8*(lane>=16)
    int row = m0 + v + 8 * hl;
    float cp = cs[(size_t)idx[row] * F_DIM + col];
    float gi = acc[0][v] + bias[0];
    float gf = acc[1][v] + bias[1];
    float gg = acc[2][v] + bias[2];
    float go = acc[3][v] + bias[3];
    float cn = sigmoidf_(gf) * cp + sigmoidf_(gi) * tanhf(gg);
    float hn = sigmoidf_(go) * tanhf(cn);
    h_new[(size_t)row * F_DIM + col] = f2bf(hn);
  }
}

// ---------------- generic GEMM: C[m,n] = sum_k A[m,k]*W[n,k] (+bias), bf16 in ----------------
template <bool BF16_OUT>
__global__ __launch_bounds__(256) void gemm_kernel(
    const __bf16* __restrict__ A, const __bf16* __restrict__ W,
    const float* __restrict__ bias, float* __restrict__ Cf, __bf16* __restrict__ Cb) {
  const int lane  = threadIdx.x & 31;
  const int wave  = threadIdx.x >> 5;
  const int patch = blockIdx.x * 8 + wave;
  const int NT = F_DIM / 16;
  const int m0 = (patch / NT) * 16;
  const int n0 = (patch % NT) * 16;
  const int r  = lane & 15, hl = lane >> 4;

  const __bf16* arow = A + (size_t)(m0 + r) * F_DIM;
  v8f acc = v8f{};
  for (int k0 = 0; k0 < F_DIM; k0 += 32) {
    v16bf a = load_a_frag(arow, k0, hl);
    v16bf b = load_b_frag(W, n0 + r, k0, hl);
    acc = wmma_bf16(a, b, acc);
  }
  const int col = n0 + r;
  float bs = bias ? bias[col] : 0.0f;
#pragma unroll
  for (int v = 0; v < 8; ++v) {
    int row = m0 + v + 8 * hl;
    float val = acc[v] + bs;
    if (BF16_OUT) Cb[(size_t)row * F_DIM + col] = f2bf(val);
    else          Cf[(size_t)row * F_DIM + col] = val;
  }
}

// ---------------- K4: per-row memory scatter/decay + tiny attention -> wmem (bf16) ---------
__global__ __launch_bounds__(256) void attn_kernel(
    const int* __restrict__ idx, const float* __restrict__ x,
    const float* __restrict__ tm, const int* __restrict__ mptr,
    const __bf16* __restrict__ qb, const float* __restrict__ qk,
    const float* __restrict__ bk, __bf16* __restrict__ wmem) {
  const int lane = threadIdx.x & 31;
  const int b    = blockIdx.x * 8 + (threadIdx.x >> 5);
  const int nid  = idx[b];
  const int ptr  = mptr[nid];
  const int off  = lane * 8;                 // 8 contiguous features per lane

  const float* mrow = tm + (size_t)nid * M_SLOTS * F_DIM;
  float mv[M_SLOTS][8];
#pragma unroll
  for (int m = 0; m < M_SLOTS; ++m) {
    const float4* p = (const float4*)(mrow + (size_t)m * F_DIM + off);
    float4 a = p[0], c = p[1];
    mv[m][0]=a.x; mv[m][1]=a.y; mv[m][2]=a.z; mv[m][3]=a.w;
    mv[m][4]=c.x; mv[m][5]=c.y; mv[m][6]=c.z; mv[m][7]=c.w;
  }
  // scatter new features + decay previous slot (unique indices => purely local)
  const float4* xp = (const float4*)(x + (size_t)b * F_DIM + off);
  float4 xa = xp[0], xc = xp[1];
  float xv[8] = {xa.x, xa.y, xa.z, xa.w, xc.x, xc.y, xc.z, xc.w};
#pragma unroll
  for (int m = 0; m < M_SLOTS; ++m) {
    bool isptr  = (m == ptr);
    bool isprev = (ptr > 0) && (m == ptr - 1);
#pragma unroll
    for (int j = 0; j < 8; ++j)
      mv[m][j] = isptr ? xv[j] : (isprev ? mv[m][j] * DECAYF : mv[m][j]);
  }
  // scores_m = <qk_b, mem_m>;  bias term <q_b, bk>
  const float4* qkp = (const float4*)(qk + (size_t)b * F_DIM + off);
  float4 k0 = qkp[0], k1 = qkp[1];
  float qkv[8] = {k0.x,k0.y,k0.z,k0.w,k1.x,k1.y,k1.z,k1.w};
  union { uint4 u; __bf16 h[8]; } qa;
  qa.u = *(const uint4*)(qb + (size_t)b * F_DIM + off);
  const float4* bkp = (const float4*)(bk + off);
  float4 b0 = bkp[0], b1 = bkp[1];
  float bkv[8] = {b0.x,b0.y,b0.z,b0.w,b1.x,b1.y,b1.z,b1.w};
  float qbk = 0.0f;
#pragma unroll
  for (int j = 0; j < 8; ++j) qbk += (float)qa.h[j] * bkv[j];
  float s[M_SLOTS];
#pragma unroll
  for (int m = 0; m < M_SLOTS; ++m) {
    float acc = 0.0f;
#pragma unroll
    for (int j = 0; j < 8; ++j) acc += qkv[j] * mv[m][j];
    s[m] = acc;
  }
  // wave32 reductions
#pragma unroll
  for (int o = 16; o >= 1; o >>= 1) {
    qbk += __shfl_xor(qbk, o, 32);
#pragma unroll
    for (int m = 0; m < M_SLOTS; ++m) s[m] += __shfl_xor(s[m], o, 32);
  }
  // softmax over M (scale 1/sqrt(256) = 1/16)
  const float inv = 1.0f / 16.0f;
  float mx = -3.4e38f;
#pragma unroll
  for (int m = 0; m < M_SLOTS; ++m) { s[m] = (s[m] + qbk) * inv; mx = fmaxf(mx, s[m]); }
  float den = 0.0f;
#pragma unroll
  for (int m = 0; m < M_SLOTS; ++m) { s[m] = __expf(s[m] - mx); den += s[m]; }
  const float rden = 1.0f / den;
  float outp[8] = {0,0,0,0,0,0,0,0};
#pragma unroll
  for (int m = 0; m < M_SLOTS; ++m) {
    float w = s[m] * rden;
#pragma unroll
    for (int j = 0; j < 8; ++j) outp[j] += w * mv[m][j];
  }
  union { uint4 u; __bf16 h[8]; } ov;
#pragma unroll
  for (int j = 0; j < 8; ++j) ov.h[j] = f2bf(outp[j]);
  *(uint4*)(wmem + (size_t)b * F_DIM + off) = ov.u;
}

// ---------------- host side ----------------
extern "C" void kernel_launch(void* const* d_in, const int* in_sizes, int n_in,
                              void* d_out, int out_size, void* d_ws, size_t ws_size,
                              hipStream_t stream) {
  const int*   idx  = (const int*)  d_in[0];
  const float* x    = (const float*)d_in[1];
  const float* hs   = (const float*)d_in[2];
  const float* cs   = (const float*)d_in[3];
  const float* tm   = (const float*)d_in[4];
  const int*   mptr = (const int*)  d_in[5];
  const float* W_ih = (const float*)d_in[6];
  const float* W_hh = (const float*)d_in[7];
  const float* b_ih = (const float*)d_in[8];
  const float* b_hh = (const float*)d_in[9];
  const float* Wq   = (const float*)d_in[10];
  const float* bq   = (const float*)d_in[11];
  const float* Wk   = (const float*)d_in[12];
  const float* bk   = (const float*)d_in[13];
  const float* Wv   = (const float*)d_in[14];
  const float* bv   = (const float*)d_in[15];
  float* out = (float*)d_out;

  // ---- workspace carve-up (bf16 buffers + one f32 buffer) ----
  const size_t BF = (size_t)B_ROWS * F_DIM;          // 8,388,608
  const size_t WG = (size_t)4 * F_DIM * F_DIM;       // 262,144 (gate weights)
  const size_t WS = (size_t)F_DIM * F_DIM;           // 65,536
  char* p = (char*)d_ws;
  __bf16* xb    = (__bf16*)p; p += BF * 2;           // x (bf16)
  __bf16* hgb   = (__bf16*)p; p += BF * 2;           // gathered h_prev (bf16)
  __bf16* hnb   = (__bf16*)p; p += BF * 2;           // h_new (bf16)
  __bf16* qbuf  = (__bf16*)p; p += BF * 2;           // q (bf16)
  __bf16* wmemb = (__bf16*)p; p += BF * 2;           // weighted memory (bf16)
  float*  qkf   = (float*)p;  p += BF * 4;           // qk (f32)
  __bf16* wihb  = (__bf16*)p; p += WG * 2;
  __bf16* whhb  = (__bf16*)p; p += WG * 2;
  __bf16* wqb   = (__bf16*)p; p += WS * 2;
  __bf16* wktb  = (__bf16*)p; p += WS * 2;
  __bf16* wvb   = (__bf16*)p; p += WS * 2;

  const dim3 blk(256);
  const int patches   = (B_ROWS / 16) * (F_DIM / 16);
  const dim3 gemm_grid(patches / 8);
  const dim3 row_grid(B_ROWS / 8);
  const dim3 cvt_grid(2048);

  // ---- prep: one-time conversions (RNE), gathers, Wk transpose ----
  cvt_kernel<<<cvt_grid, blk, 0, stream>>>(x,    xb,   (int)(BF / 4));
  cvt_kernel<<<cvt_grid, blk, 0, stream>>>(W_ih, wihb, (int)(WG / 4));
  cvt_kernel<<<cvt_grid, blk, 0, stream>>>(W_hh, whhb, (int)(WG / 4));
  cvt_kernel<<<cvt_grid, blk, 0, stream>>>(Wq,   wqb,  (int)(WS / 4));
  cvt_kernel<<<cvt_grid, blk, 0, stream>>>(Wv,   wvb,  (int)(WS / 4));
  gather_cvt_kernel<<<dim3(BF / 4 / 256), blk, 0, stream>>>(hs, idx, hgb);
  tpose_cvt_kernel<<<dim3(WS / 256), blk, 0, stream>>>(Wk, wktb);

  // K1: gates GEMM (bf16 WMMA, f32 accum) fused with LSTM nonlinearity -> h_new
  lstm_kernel<<<gemm_grid, blk, 0, stream>>>(idx, xb, hgb, cs, wihb, whhb, b_ih, b_hh, hnb);
  // K2: q = h_new @ Wq^T + bq   (bf16 out)
  gemm_kernel<true ><<<gemm_grid, blk, 0, stream>>>(hnb, wqb, bq, nullptr, qbuf);
  // K3: qk = q @ Wk             (f32 out, W pre-transposed)
  gemm_kernel<false><<<gemm_grid, blk, 0, stream>>>(qbuf, wktb, nullptr, qkf, nullptr);
  // K4: memory scatter/decay + attention -> wmem (bf16)
  attn_kernel<<<row_grid, blk, 0, stream>>>(idx, x, tm, mptr, qbuf, qkf, bk, wmemb);
  // K5: context = wmem @ Wv^T + bv  (f32 out -> d_out)
  gemm_kernel<false><<<gemm_grid, blk, 0, stream>>>(wmemb, wvb, bv, out, nullptr);
}
